// gnn_70781061038728
// MI455X (gfx1250) — compile-verified, matrix operand order checked by reading, not run on hardware
//
#include <hip/hip_runtime.h>
#include <hip/hip_bf16.h>
#include <math.h>

// ---------------------------------------------------------------------------
// Problem constants (match reference)
// ---------------------------------------------------------------------------
#define FIN   56
#define CCH   128      // C
#define HH    2        // heads
#define LL    3        // layers
#define GG    800      // graphs
#define NEGS  0.2f

typedef __attribute__((ext_vector_type(16))) __bf16 v16bf;
typedef __attribute__((ext_vector_type(8)))  float  v8f;
typedef int v4i32 __attribute__((vector_size(16)));

union BF16x16 { uint4 q[2]; v16bf v; __bf16 e[16]; };
union F32x8   { v8f   v; float  e[8]; };

// async-LDS availability (device pass only)
#if defined(__AMDGCN__) && __has_builtin(__builtin_amdgcn_global_load_async_to_lds_b128)
#define USE_ASYNC_LDS 1
#else
#define USE_ASYNC_LDS 0
#endif

#define AS1 __attribute__((address_space(1)))
#define AS3 __attribute__((address_space(3)))

// ---------------------------------------------------------------------------
// Packed-B WMMA GEMM:  C[M x Nn] = A[M x K] (row-major bf16) * B (packed bf16)
//
// Bpack layout: for N-group g (64 cols), k-step ks, tile t(0..3), lane(0..31):
// 16 contiguous bf16 = exactly the lane's 16x16x32 B fragment.
// Block = 256 threads = 8 waves; all waves share one N-group panel staged in
// LDS (async-to-LDS if available); each wave computes a 16x64 strip
// (4 accumulators reusing one A fragment), K fully unrolled.
// ---------------------------------------------------------------------------
template <int K>
__global__ void __launch_bounds__(256)
wmma_gemm_tiled_kernel(const __bf16* __restrict__ A,
                       const __bf16* __restrict__ Bpack,
                       float* __restrict__ Cmat,
                       int M, int Nn) {
    constexpr int KS = K / 32;
    constexpr int PANEL_ELEMS = KS * 4 * 32 * 16;        // bf16 per group panel
    __shared__ __align__(16) __bf16 panel[PANEL_ELEMS];

    const int ngroups = Nn >> 6;
    const int gid = blockIdx.x % ngroups;                // N-group (64 cols)
    const int mb  = blockIdx.x / ngroups;                // M block (128 rows)
    const int tid = threadIdx.x;

    // ---- stage B panel into LDS (16B chunks, 256 threads) ----
    const __bf16* gsrc = Bpack + (size_t)gid * PANEL_ELEMS;
    constexpr int CHUNKS = PANEL_ELEMS / 8;              // 16-byte chunks
#if USE_ASYNC_LDS
    for (int i = tid; i < CHUNKS; i += 256)
        __builtin_amdgcn_global_load_async_to_lds_b128(
            (AS1 v4i32*)(gsrc + (size_t)i * 8),
            (AS3 v4i32*)(panel + (size_t)i * 8), 0, 0);
#if __has_builtin(__builtin_amdgcn_s_wait_asynccnt)
    __builtin_amdgcn_s_wait_asynccnt(0);
#else
    asm volatile("s_wait_asynccnt 0x0" ::: "memory");
#endif
#else
    for (int i = tid; i < CHUNKS; i += 256)
        *(uint4*)(panel + (size_t)i * 8) = *(const uint4*)(gsrc + (size_t)i * 8);
#endif
    __syncthreads();

    // ---- per-wave 16x64 strip ----
    const int wv   = tid >> 5;
    const int lane = tid & 31;
    const int half = lane >> 4;
    const int r16  = lane & 15;
    const int mt   = mb * 8 + wv;                        // 16-row tile index
    const bool ok  = (mt * 16) < M;

    if (ok) {
        F32x8 acc[4];
#pragma unroll
        for (int t = 0; t < 4; ++t)
            acc[t].v = (v8f){0.f,0.f,0.f,0.f,0.f,0.f,0.f,0.f};

        const __bf16* Arow = A + (size_t)(mt * 16 + r16) * K;
        __builtin_prefetch(Arow, 0, 0);

#pragma unroll
        for (int ks = 0; ks < KS; ++ks) {
            // A fragment: two contiguous 16B chunks (layout per ISA 7.12.2)
            BF16x16 a;
            const uint4* ap = (const uint4*)(Arow + ks * 32 + half * 8);
            a.q[0] = ap[0];          // K = k0      .. k0+7   (this half)
            a.q[1] = ap[2];          // K = k0 + 16 .. k0+23  (this half)
#pragma unroll
            for (int t = 0; t < 4; ++t) {
                BF16x16 b;
                const uint4* bp =
                    (const uint4*)&panel[((ks * 4 + t) * 32 + lane) * 16];
                b.q[0] = bp[0];
                b.q[1] = bp[1];
                acc[t].v = __builtin_amdgcn_wmma_f32_16x16x32_bf16(
                    false, a.v, false, b.v, (short)0, acc[t].v, false, false);
            }
        }
        // ---- store: C/D layout: VGPR r -> row r + 8*half ----
#pragma unroll
        for (int t = 0; t < 4; ++t) {
            int n = gid * 64 + t * 16 + r16;
#pragma unroll
            for (int r = 0; r < 8; ++r)
                Cmat[(size_t)(mt * 16 + r + half * 8) * Nn + n] = acc[t].e[r];
        }
    }
}

// ---------------------------------------------------------------------------
// Repack a row-major f32 KxNn weight matrix into WMMA-fragment order (bf16).
// kin < K zero-pads the K dimension (embedding: 56 -> 64).
// ---------------------------------------------------------------------------
__global__ void repack_B_kernel(const float* __restrict__ B, __bf16* __restrict__ P,
                                int K, int Nn, int kin) {
    int idx = blockIdx.x * blockDim.x + threadIdx.x;
    if (idx >= K * Nn) return;
    int e    = idx & 15;
    int lane = (idx >> 4) & 31;
    int t    = (idx >> 9) & 3;
    int rest = idx >> 11;
    int ksn  = K >> 5;
    int ks   = rest % ksn;
    int g    = rest / ksn;
    int half = lane >> 4, r16 = lane & 15;
    int v = e >> 1;
    int k = ks * 32 + ((v >= 4) ? 16 : 0) + (v & 3) * 2 + half * 8 + (e & 1);
    int n = g * 64 + t * 16 + r16;
    P[idx] = (__bf16)((k < kin) ? B[(size_t)k * Nn + n] : 0.f);
}

// ---------------------------------------------------------------------------
// Utility kernels
// ---------------------------------------------------------------------------
__global__ void fill_f32_kernel(float* p, float v, size_t n) {
    size_t i = blockIdx.x * (size_t)blockDim.x + threadIdx.x;
    if (i < n) p[i] = v;
}

__global__ void f32_to_bf16_kernel(const float* __restrict__ s, __bf16* __restrict__ d, size_t n) {
    size_t i = blockIdx.x * (size_t)blockDim.x + threadIdx.x;
    if (i < n) d[i] = (__bf16)s[i];
}

// pad columns: (rows x kin) f32 -> (rows x kout) bf16, zeros beyond kin
__global__ void pad_cols_bf16_kernel(const float* __restrict__ s, __bf16* __restrict__ d,
                                     int rows, int kin, int kout) {
    size_t i = blockIdx.x * (size_t)blockDim.x + threadIdx.x;
    if (i >= (size_t)rows * kout) return;
    int r = (int)(i / kout), c = (int)(i % kout);
    d[i] = (__bf16)((c < kin) ? s[(size_t)r * kin + c] : 0.f);
}

// a1 = ea[:,0]; a2 = exp(-(ea[:,1]-mu)^2/dev) + a1
__global__ void edge_attr_kernel(const float* __restrict__ ea, const float* __restrict__ mu,
                                 const float* __restrict__ dev, float* __restrict__ a1,
                                 float* __restrict__ a2, int ne) {
    int e = blockIdx.x * blockDim.x + threadIdx.x;
    if (e >= ne) return;
    float v1 = ea[(size_t)e * 2 + 0];
    float t  = ea[(size_t)e * 2 + 1] - mu[0];
    a1[e] = v1;
    a2[e] = __expf(-t * t / dev[0]) + v1;
}

// ls/ld[n,h] = sum_c hh[n,h,c] * att_{s,d}[h,c]
__global__ void node_logits_kernel(const float* __restrict__ hh,
                                   const float* __restrict__ att_s,
                                   const float* __restrict__ att_d,
                                   float* __restrict__ ls, float* __restrict__ ld,
                                   int n_nodes) {
    int idx = blockIdx.x * blockDim.x + threadIdx.x;   // n*2 + h
    if (idx >= n_nodes * HH) return;
    int hd = idx & 1, n = idx >> 1;
    const float* row = hh + (size_t)n * (HH * CCH) + hd * CCH;
    float s = 0.f, d = 0.f;
    for (int c = 0; c < CCH; ++c) {
        float v = row[c];
        s += v * att_s[hd * CCH + c];
        d += v * att_d[hd * CCH + c];
    }
    ls[idx] = s;
    ld[idx] = d;
}

// se[h] = dot(We[h], att_e[h])   (le[e,h] = a[e]*se[h])
__global__ void se_kernel(const float* __restrict__ We_k, const float* __restrict__ att_e_k,
                          float* __restrict__ se) {
    int hd = threadIdx.x;
    if (hd >= HH) return;
    float s = 0.f;
    for (int c = 0; c < CCH; ++c) s += We_k[hd * CCH + c] * att_e_k[hd * CCH + c];
    se[hd] = s;
}

// float atomic max via monotonic integer-bits trick (init must be -inf)
__device__ __forceinline__ void atomicMaxFloat(float* addr, float val) {
    if (val >= 0.f)
        atomicMax((int*)addr, __float_as_int(val));
    else
        atomicMin((unsigned int*)addr, (unsigned int)__float_as_int(val));
}

// pass A: logit = leaky_relu(ls[src]+ld[dst]+a*se); segment max into mbuf
__global__ void edge_logit_max_kernel(const int* __restrict__ src, const int* __restrict__ dstv,
                                      const float* __restrict__ a, const float* __restrict__ ls,
                                      const float* __restrict__ ld, const float* __restrict__ se,
                                      float* __restrict__ logit, float* __restrict__ mbuf, int ne) {
    int e = blockIdx.x * blockDim.x + threadIdx.x;
    if (e >= ne) return;
    int s = src[e], d = dstv[e];
    float av = a[e];
#pragma unroll
    for (int hd = 0; hd < HH; ++hd) {
        float lg = ls[s * HH + hd] + ld[d * HH + hd] + av * se[hd];
        lg = lg > 0.f ? lg : NEGS * lg;
        logit[(size_t)e * HH + hd] = lg;
        atomicMaxFloat(&mbuf[d * HH + hd], lg);
    }
}

// pass B: w = exp(logit - m[dst]); segment sum; logit <- w
__global__ void edge_exp_sum_kernel(const int* __restrict__ dstv, float* __restrict__ logit,
                                    const float* __restrict__ mbuf, float* __restrict__ sbuf,
                                    int ne) {
    int e = blockIdx.x * blockDim.x + threadIdx.x;
    if (e >= ne) return;
    int d = dstv[e];
#pragma unroll
    for (int hd = 0; hd < HH; ++hd) {
        float w = __expf(logit[(size_t)e * HH + hd] - mbuf[d * HH + hd]);
        logit[(size_t)e * HH + hd] = w;
        atomicAdd(&sbuf[d * HH + hd], w);
    }
}

// pass C: out[dst,h,c] += alpha * hh[src,h,c]   (one wave per edge, 256 f32)
__global__ void edge_aggregate_kernel(const int* __restrict__ src, const int* __restrict__ dstv,
                                      const float* __restrict__ w, const float* __restrict__ sbuf,
                                      const float* __restrict__ hh, float* __restrict__ out, int ne) {
    int wid  = (blockIdx.x * blockDim.x + threadIdx.x) >> 5;
    int lane = threadIdx.x & 31;
    if (wid >= ne) return;
    int s = src[wid], d = dstv[wid];
    float al0 = w[(size_t)wid * HH + 0] / (sbuf[d * HH + 0] + 1e-16f);
    float al1 = w[(size_t)wid * HH + 1] / (sbuf[d * HH + 1] + 1e-16f);
    const float* hrow = hh + (size_t)s * (HH * CCH);
    float* orow = out + (size_t)d * (HH * CCH);
#pragma unroll
    for (int i = 0; i < 8; ++i) {
        int j = lane + i * 32;                 // 0..255 = h*128 + c
        float al = (j < CCH) ? al0 : al1;
        atomicAdd(&orow[j], al * hrow[j]);
    }
}

// Gate + layer combine:  h <- gate(h,out2) - gate(h,out1)    (block = node)
__global__ void gate_combine_kernel(float* __restrict__ h, const float* __restrict__ out1,
                                    const float* __restrict__ out2, const float* __restrict__ bg,
                                    const float* __restrict__ gw, const float* __restrict__ gb) {
    __shared__ float r1[CCH], r2[CCH];
    int n = blockIdx.x;
    int c = threadIdx.x;
    float o1 = 0.5f * (out1[(size_t)n * 256 + c] + out1[(size_t)n * 256 + CCH + c]) + bg[c];
    float o2 = 0.5f * (out2[(size_t)n * 256 + c] + out2[(size_t)n * 256 + CCH + c]) + bg[c];
    float hv = h[(size_t)n * CCH + c];
    r1[c] = gw[c] * hv + gw[CCH + c] * o1;
    r2[c] = gw[c] * hv + gw[CCH + c] * o2;
    __syncthreads();
    for (int sft = CCH / 2; sft > 0; sft >>= 1) {
        if (c < sft) { r1[c] += r1[c + sft]; r2[c] += r2[c + sft]; }
        __syncthreads();
    }
    float z1 = 1.f / (1.f + __expf(-(r1[0] + gb[0])));
    float z2 = 1.f / (1.f + __expf(-(r2[0] + gb[0])));
    float h1 = z1 * hv + (1.f - z1) * fmaxf(o1, 0.f);
    float h2 = z2 * hv + (1.f - z2) * fmaxf(o2, 0.f);
    h[(size_t)n * CCH + c] = h2 - h1;
}

// Pooling: pooled[g,0:128] += h*valid0 ; pooled[g,128:256] += h*valid1
__global__ void pool_kernel(const float* __restrict__ h, const float* __restrict__ valid,
                            const int* __restrict__ batch, float* __restrict__ pooled,
                            int n_nodes) {
    int idx = blockIdx.x * blockDim.x + threadIdx.x;   // n*128 + c
    if (idx >= n_nodes * CCH) return;
    int n = idx >> 7, c = idx & 127;
    int g = batch[n];
    float hv = h[idx];
    atomicAdd(&pooled[(size_t)g * 256 + c],       hv * valid[n * 2 + 0]);
    atomicAdd(&pooled[(size_t)g * 256 + CCH + c], hv * valid[n * 2 + 1]);
}

// tiny MLP dense: out = [relu](in @ W + b)
__global__ void dense_kernel(const float* __restrict__ in, const float* __restrict__ W,
                             const float* __restrict__ b, float* __restrict__ out,
                             int M, int K, int Nn, int do_relu) {
    int idx = blockIdx.x * blockDim.x + threadIdx.x;
    if (idx >= M * Nn) return;
    int m = idx / Nn, j = idx % Nn;
    float s = b[j];
    for (int i = 0; i < K; ++i) s += in[(size_t)m * K + i] * W[(size_t)i * Nn + j];
    out[idx] = do_relu ? fmaxf(s, 0.f) : s;
}

__global__ void final_kernel(const float* __restrict__ o, float* __restrict__ out, int g) {
    int i = blockIdx.x * blockDim.x + threadIdx.x;
    if (i >= g) return;
    out[i * 2 + 0] = o[i * 2 + 0];
    out[i * 2 + 1] = __expf(o[i * 2 + 1]);
}

// ---------------------------------------------------------------------------
// Host orchestration
// ---------------------------------------------------------------------------
extern "C" void kernel_launch(void* const* d_in, const int* in_sizes, int n_in,
                              void* d_out, int out_size, void* d_ws, size_t ws_size,
                              hipStream_t stream) {
    const float* x       = (const float*)d_in[0];
    const int*   eidx    = (const int*)  d_in[1];
    const float* ea      = (const float*)d_in[2];
    const float* valid   = (const float*)d_in[3];
    const int*   batch   = (const int*)  d_in[4];
    const float* mu      = (const float*)d_in[5];
    const float* devp    = (const float*)d_in[6];
    const float* W_emb   = (const float*)d_in[7];
    const float* Wg      = (const float*)d_in[8];
    const float* att_src = (const float*)d_in[9];
    const float* att_dst = (const float*)d_in[10];
    const float* We      = (const float*)d_in[11];
    const float* att_e   = (const float*)d_in[12];
    const float* bg      = (const float*)d_in[13];
    const float* gw      = (const float*)d_in[14];
    const float* gb      = (const float*)d_in[15];
    const float* fc_w0   = (const float*)d_in[16];
    const float* fc_b0   = (const float*)d_in[17];
    const float* fc_w1   = (const float*)d_in[18];
    const float* fc_b1   = (const float*)d_in[19];
    const float* fc_w2   = (const float*)d_in[20];
    const float* fc_b2   = (const float*)d_in[21];
    const float* fc_w3   = (const float*)d_in[22];
    const float* fc_b3   = (const float*)d_in[23];

    const int NN = in_sizes[0] / FIN;      // 40000
    const int NE = in_sizes[1] / 2;        // 500000
    const int* src = eidx;
    const int* dst = eidx + NE;

    // workspace carve
    size_t off = 0;
    auto alloc = [&](size_t bytes) -> void* {
        void* p = (char*)d_ws + off;
        off += (bytes + 255) & ~(size_t)255;
        return p;
    };
    float*  h        = (float*) alloc((size_t)NN * CCH * 4);
    __bf16* h_bf     = (__bf16*)alloc((size_t)NN * CCH * 2);
    __bf16* xb       = (__bf16*)alloc((size_t)NN * 64 * 2);
    __bf16* wembpack = (__bf16*)alloc((size_t)64 * CCH * 2);
    __bf16* wgpack   = (__bf16*)alloc((size_t)LL * CCH * 256 * 2);
    float*  hh       = (float*) alloc((size_t)NN * 256 * 4);
    float*  out1     = (float*) alloc((size_t)NN * 256 * 4);
    float*  out2     = (float*) alloc((size_t)NN * 256 * 4);
    float*  ls       = (float*) alloc((size_t)NN * HH * 4);
    float*  ld       = (float*) alloc((size_t)NN * HH * 4);
    float*  mbuf     = (float*) alloc((size_t)NN * HH * 4);
    float*  sbuf     = (float*) alloc((size_t)NN * HH * 4);
    float*  wlog     = (float*) alloc((size_t)NE * HH * 4);
    float*  a1       = (float*) alloc((size_t)NE * 4);
    float*  a2       = (float*) alloc((size_t)NE * 4);
    float*  se       = (float*) alloc(HH * 4);
    float*  pooled   = (float*) alloc((size_t)GG * 256 * 4);
    float*  t0       = (float*) alloc((size_t)GG * 128 * 4);
    float*  t1       = (float*) alloc((size_t)GG * 128 * 4);
    float*  t2       = (float*) alloc((size_t)GG * 2 * 4);

    auto cdiv = [](long a, long b) { return (a + b - 1) / b; };
    const int T = 256;
    const int Mblocks = (int)cdiv(NN, 128);

    // --- setup: conversions, repacks, edge attrs ---
    pad_cols_bf16_kernel<<<cdiv((size_t)NN * 64, T), T, 0, stream>>>(x, xb, NN, FIN, 64);
    repack_B_kernel<<<cdiv(64 * CCH, T), T, 0, stream>>>(W_emb, wembpack, 64, CCH, FIN);
    for (int k = 0; k < LL; ++k)
        repack_B_kernel<<<cdiv(CCH * 256, T), T, 0, stream>>>(
            Wg + (size_t)k * CCH * 256, wgpack + (size_t)k * CCH * 256, CCH, 256, CCH);
    edge_attr_kernel<<<cdiv(NE, T), T, 0, stream>>>(ea, mu, devp, a1, a2, NE);

    // --- embedding GEMM: h = x @ W_emb  (K padded 56->64, Nn=128 -> 2 groups) ---
    wmma_gemm_tiled_kernel<64><<<Mblocks * 2, 256, 0, stream>>>(xb, wembpack, h, NN, CCH);
    f32_to_bf16_kernel<<<cdiv((size_t)NN * CCH, T), T, 0, stream>>>(h, h_bf, (size_t)NN * CCH);

    // --- layers ---
    for (int k = 0; k < LL; ++k) {
        // hh = h @ Wg[k]  (N x 256, K=128 -> 4 groups) — dominant WMMA GEMM
        wmma_gemm_tiled_kernel<128><<<Mblocks * 4, 256, 0, stream>>>(
            h_bf, wgpack + (size_t)k * CCH * 256, hh, NN, 256);

        node_logits_kernel<<<cdiv(NN * HH, T), T, 0, stream>>>(
            hh, att_src + k * HH * CCH, att_dst + k * HH * CCH, ls, ld, NN);
        se_kernel<<<1, 32, 0, stream>>>(We + k * HH * CCH, att_e + k * HH * CCH, se);

        for (int v = 0; v < 2; ++v) {
            const float* a   = v ? a2   : a1;
            float*       out = v ? out2 : out1;
            fill_f32_kernel<<<cdiv(NN * HH, T), T, 0, stream>>>(mbuf, -__builtin_inff(),
                                                                (size_t)NN * HH);
            fill_f32_kernel<<<cdiv(NN * HH, T), T, 0, stream>>>(sbuf, 0.f, (size_t)NN * HH);
            fill_f32_kernel<<<cdiv((size_t)NN * 256, T), T, 0, stream>>>(out, 0.f,
                                                                         (size_t)NN * 256);
            edge_logit_max_kernel<<<cdiv(NE, T), T, 0, stream>>>(src, dst, a, ls, ld, se,
                                                                 wlog, mbuf, NE);
            edge_exp_sum_kernel<<<cdiv(NE, T), T, 0, stream>>>(dst, wlog, mbuf, sbuf, NE);
            edge_aggregate_kernel<<<cdiv((long)NE * 32, T), T, 0, stream>>>(
                src, dst, wlog, sbuf, hh, out, NE);
        }
        gate_combine_kernel<<<NN, CCH, 0, stream>>>(h, out1, out2, bg + k * CCH,
                                                    gw + k * 2 * CCH, gb + k);
        f32_to_bf16_kernel<<<cdiv((size_t)NN * CCH, T), T, 0, stream>>>(h, h_bf,
                                                                        (size_t)NN * CCH);
    }

    // --- pooling + MLP head ---
    fill_f32_kernel<<<cdiv((size_t)GG * 256, T), T, 0, stream>>>(pooled, 0.f, (size_t)GG * 256);
    pool_kernel<<<cdiv((size_t)NN * CCH, T), T, 0, stream>>>(h, valid, batch, pooled, NN);

    dense_kernel<<<cdiv(GG * 128, T), T, 0, stream>>>(pooled, fc_w0, fc_b0, t0, GG, 256, 128, 1);
    dense_kernel<<<cdiv(GG * 128, T), T, 0, stream>>>(t0, fc_w1, fc_b1, t1, GG, 128, 128, 1);
    dense_kernel<<<cdiv(GG * 128, T), T, 0, stream>>>(t1, fc_w2, fc_b2, t0, GG, 128, 128, 1);
    dense_kernel<<<cdiv(GG * 2, T), T, 0, stream>>>(t0, fc_w3, fc_b3, t2, GG, 128, 2, 0);
    final_kernel<<<cdiv(GG, T), T, 0, stream>>>(t2, (float*)d_out, GG);
}
